// TimeMix_71940702208372
// MI455X (gfx1250) — compile-verified, matrix operand order checked by reading, not run on hardware
//
#include <hip/hip_runtime.h>
#include <cstdint>

// ---------------------------------------------------------------------------
// RWKV TimeMix for MI455X (gfx1250, wave32).
//   T = C = 2048.  4x bf16 WMMA GEMMs (f32 accum, 2048^3 each) + chunked wkv.
//   GEMM: 128x128 block tile, BK=32, double-buffered LDS filled with
//   GLOBAL_LOAD_ASYNC_TO_LDS_B128 (ASYNCcnt pipeline), 8 waves, each wave a
//   64x32 tile = 4x2 v_wmma_f32_16x16x32_bf16 accumulators.
// Workspace: ~105 MB (bf16 W^T x4, bf16 activations x3, f32 k/v/r, scan state)
// ---------------------------------------------------------------------------

typedef __attribute__((ext_vector_type(16))) __bf16 v16bf;
typedef __attribute__((ext_vector_type(8)))  __bf16 v8bf;
typedef __attribute__((ext_vector_type(8)))  float  v8f;

#define Tdim 2048
#define Cdim 2048
#define NCH  16
#define CHL  (Tdim/NCH)

#define GM 2048
#define GN 2048
#define GK 2048
#define BM 128
#define BN 128
#define BK 32
#define NIT (GK / BK)    // 64 K-stages
#define LDSS 40          // bf16 row stride (32 + 8 pad) -> 80B, 16B aligned

__device__ __forceinline__ __bf16 f2bf(float f) {
    unsigned int u = __builtin_bit_cast(unsigned int, f);
    unsigned int r = (u + 0x7FFFu + ((u >> 16) & 1u)) >> 16;
    unsigned short s = (unsigned short)r;
    return __builtin_bit_cast(__bf16, s);
}

// async 16B global -> LDS, tracked by ASYNCcnt
#define ASYNC_LD16(lds_off, gaddr)                                        \
    asm volatile("global_load_async_to_lds_b128 %0, %1, off"              \
                 :: "v"(lds_off), "v"(gaddr) : "memory")

// -------------------- elementwise time-shift mixing -> bf16 -----------------
__global__ __launch_bounds__(256) void prep_mix(
    const float* __restrict__ x,
    const float* __restrict__ tmk, const float* __restrict__ tmv,
    const float* __restrict__ tmr,
    __bf16* __restrict__ xk, __bf16* __restrict__ xv, __bf16* __restrict__ xr)
{
    int idx = blockIdx.x * blockDim.x + threadIdx.x;
    if (idx >= Tdim * Cdim) return;
    int c = idx & (Cdim - 1);
    float cur  = x[idx];
    float prev = (idx >= Cdim) ? x[idx - Cdim] : 0.0f;
    float mk = tmk[c], mv = tmv[c], mr = tmr[c];
    xk[idx] = f2bf(cur * mk + prev * (1.0f - mk));
    xv[idx] = f2bf(cur * mv + prev * (1.0f - mv));
    xr[idx] = f2bf(cur * mr + prev * (1.0f - mr));
}

// -------------------- weight convert + transpose -> bf16 [N][K] -------------
__global__ __launch_bounds__(256) void cvt_w_t(
    const float* __restrict__ w0, const float* __restrict__ w1,
    const float* __restrict__ w2, const float* __restrict__ w3,
    __bf16* __restrict__ out)
{
    const float* srcs[4] = { w0, w1, w2, w3 };
    const float* w = srcs[blockIdx.y];
    __bf16* o = out + (size_t)blockIdx.y * Cdim * Cdim;
    int idx = blockIdx.x * blockDim.x + threadIdx.x;
    if (idx >= Cdim * Cdim) return;
    int n = idx >> 11;
    int k = idx & (Cdim - 1);
    o[idx] = f2bf(w[(size_t)k * Cdim + n]);
}

// -------------------- bf16 WMMA GEMM: C = A[GM,GK] * Bt[GN,GK]^T ------------
__device__ __forceinline__ v16bf load_frag(const __bf16* base, int row, int g) {
    union { v16bf v; v8bf h[2]; } u;
    const __bf16* p = base + row * LDSS;
    u.h[0] = *(const v8bf*)(p + g * 8);        // K 0..7   / 8..15
    u.h[1] = *(const v8bf*)(p + 16 + g * 8);   // K 16..23 / 24..31
    return u.v;
}

__global__ __launch_bounds__(256) void gemm_bf16(
    const __bf16* __restrict__ A,   // [GM,GK] row-major
    const __bf16* __restrict__ Bt,  // [GN,GK] row-major (B transposed)
    float* __restrict__ C)          // [GM,GN]
{
    __shared__ __bf16 As[2][BM * LDSS];
    __shared__ __bf16 Bs[2][BN * LDSS];

    const int m0 = blockIdx.y * BM;
    const int n0 = blockIdx.x * BN;
    const int t  = threadIdx.x;
    const int lane = t & 31;
    const int wave = t >> 5;          // 8 waves: 2 (M) x 4 (N)
    const int mw = (wave >> 2) * 64;
    const int nw = (wave & 3) * 32;
    const int g  = lane >> 4;
    const int ln = lane & 15;

    const int lrow = t >> 2;          // 0..63
    const int lcol = (t & 3) * 8;     // 0,8,16,24 (bf16 elems)

    // per-thread global source addresses; advance 64B (= BK bf16) per stage
    unsigned long long gA0 = (unsigned long long)(uintptr_t)&A [(size_t)(m0 + lrow)      * GK + lcol];
    unsigned long long gA1 = (unsigned long long)(uintptr_t)&A [(size_t)(m0 + lrow + 64) * GK + lcol];
    unsigned long long gB0 = (unsigned long long)(uintptr_t)&Bt[(size_t)(n0 + lrow)      * GK + lcol];
    unsigned long long gB1 = (unsigned long long)(uintptr_t)&Bt[(size_t)(n0 + lrow + 64) * GK + lcol];

    // per-thread LDS destinations for both buffers (low 32 bits = LDS offset)
    uint32_t lA0[2], lA1[2], lB0[2], lB1[2];
    #pragma unroll
    for (int b = 0; b < 2; ++b) {
        lA0[b] = (uint32_t)(uintptr_t)&As[b][lrow * LDSS + lcol];
        lA1[b] = (uint32_t)(uintptr_t)&As[b][(lrow + 64) * LDSS + lcol];
        lB0[b] = (uint32_t)(uintptr_t)&Bs[b][lrow * LDSS + lcol];
        lB1[b] = (uint32_t)(uintptr_t)&Bs[b][(lrow + 64) * LDSS + lcol];
    }

    v8f acc[4][2];
    #pragma unroll
    for (int mi = 0; mi < 4; ++mi)
        #pragma unroll
        for (int ni = 0; ni < 2; ++ni)
            #pragma unroll
            for (int e = 0; e < 8; ++e) acc[mi][ni][e] = 0.0f;

    // prologue: stage 0 into buffer 0
    ASYNC_LD16(lA0[0], gA0);
    ASYNC_LD16(lA1[0], gA1);
    ASYNC_LD16(lB0[0], gB0);
    ASYNC_LD16(lB1[0], gB1);

    for (int it = 0; it < NIT; ++it) {
        const int cur = it & 1;
        if (it + 1 < NIT) {
            const int nxt = cur ^ 1;
            const unsigned long long adv = (unsigned long long)((it + 1) * (BK * 2));
            ASYNC_LD16(lA0[nxt], gA0 + adv);
            ASYNC_LD16(lA1[nxt], gA1 + adv);
            ASYNC_LD16(lB0[nxt], gB0 + adv);
            ASYNC_LD16(lB1[nxt], gB1 + adv);
            asm volatile("s_wait_asynccnt 0x4" ::: "memory");  // stage `it` landed
        } else {
            asm volatile("s_wait_asynccnt 0x0" ::: "memory");
        }
        __syncthreads();   // all waves' stage-`it` data visible

        const __bf16* a_lds = As[cur];
        const __bf16* b_lds = Bs[cur];
        v16bf af[4], bfr[2];
        #pragma unroll
        for (int mi = 0; mi < 4; ++mi) af[mi]  = load_frag(a_lds, mw + mi * 16 + ln, g);
        #pragma unroll
        for (int ni = 0; ni < 2; ++ni) bfr[ni] = load_frag(b_lds, nw + ni * 16 + ln, g);

        #pragma unroll
        for (int mi = 0; mi < 4; ++mi)
            #pragma unroll
            for (int ni = 0; ni < 2; ++ni)
                acc[mi][ni] = __builtin_amdgcn_wmma_f32_16x16x32_bf16(
                    false, af[mi], false, bfr[ni], (short)0, acc[mi][ni],
                    false, false);
        __syncthreads();   // done reading buf[cur]; it may be refilled next iter
    }

    // C/D layout: VGPR r, lanes 0-15 -> M=r,N=lane; lanes 16-31 -> M=r+8,N=ln.
    float* cptr = C + (size_t)(m0 + mw + 8 * g) * GN + (n0 + nw + ln);
    #pragma unroll
    for (int mi = 0; mi < 4; ++mi)
        #pragma unroll
        for (int r = 0; r < 8; ++r)
            #pragma unroll
            for (int ni = 0; ni < 2; ++ni)
                cptr[(size_t)(mi * 16 + r) * GN + ni * 16] = acc[mi][ni][r];
}

// -------------------- wkv chunked scan --------------------------------------
__global__ __launch_bounds__(256) void wkv_partial(
    const float* __restrict__ Kf, const float* __restrict__ Vf,
    const float* __restrict__ td,
    float* __restrict__ Pa, float* __restrict__ Pb)
{
    int c = blockIdx.x * blockDim.x + threadIdx.x;
    int j = blockIdx.y;
    float q = __expf(-__expf(td[c]));
    float d = __expf(-q);
    float a = 0.0f, b = 0.0f;
    int t0 = j * CHL;
    for (int i = 0; i < CHL; ++i) {
        size_t off = (size_t)(t0 + i) * Cdim + c;
        float ek  = __expf(Kf[off]);
        float ekv = ek * Vf[off];
        a = d * a + ekv;
        b = d * b + ek;
    }
    Pa[c * NCH + j] = a;
    Pb[c * NCH + j] = b;
}

__global__ __launch_bounds__(256) void wkv_scan(
    const float* __restrict__ Pa, const float* __restrict__ Pb,
    const float* __restrict__ td,
    float* __restrict__ Sa, float* __restrict__ Sb)
{
    int c = blockIdx.x * blockDim.x + threadIdx.x;
    float q  = __expf(-__expf(td[c]));
    float dL = __expf(-q * (float)CHL);
    float a = 0.0f, b = 0.0f;
    #pragma unroll
    for (int j = 0; j < NCH; ++j) {
        Sa[c * NCH + j] = a;
        Sb[c * NCH + j] = b;
        a = dL * a + Pa[c * NCH + j];
        b = dL * b + Pb[c * NCH + j];
    }
}

__global__ __launch_bounds__(256) void wkv_out(
    const float* __restrict__ Kf, const float* __restrict__ Vf,
    const float* __restrict__ Rf,
    const float* __restrict__ td, const float* __restrict__ tf,
    const float* __restrict__ Sa, const float* __restrict__ Sb,
    __bf16* __restrict__ out)
{
    int c = blockIdx.x * blockDim.x + threadIdx.x;
    int j = blockIdx.y;
    float q = __expf(-__expf(td[c]));
    float d = __expf(-q);
    float eu = __expf(tf[c]);
    float a = Sa[c * NCH + j];
    float b = Sb[c * NCH + j];
    int t0 = j * CHL;
    for (int i = 0; i < CHL; ++i) {
        size_t off = (size_t)(t0 + i) * Cdim + c;
        float ek  = __expf(Kf[off]);
        float ekv = ek * Vf[off];
        float num = eu * ekv + a;
        float den = eu * ek + b;
        float r   = Rf[off];
        float sr  = 1.0f / (1.0f + __expf(-r));
        out[off]  = f2bf(sr * (num / den));
        a = d * a + ekv;
        b = d * b + ek;
    }
}

// ---------------------------------------------------------------------------
extern "C" void kernel_launch(void* const* d_in, const int* in_sizes, int n_in,
                              void* d_out, int out_size, void* d_ws, size_t ws_size,
                              hipStream_t stream) {
    (void)in_sizes; (void)n_in; (void)out_size; (void)ws_size;
    const float* x   = (const float*)d_in[0];
    const float* td  = (const float*)d_in[1];
    const float* tf  = (const float*)d_in[2];
    const float* tmk = (const float*)d_in[3];
    const float* tmv = (const float*)d_in[4];
    const float* tmr = (const float*)d_in[5];
    const float* Wk  = (const float*)d_in[6];
    const float* Wv  = (const float*)d_in[7];
    const float* Wr  = (const float*)d_in[8];
    const float* Wo  = (const float*)d_in[9];

    const size_t TC = (size_t)Tdim * Cdim;
    char* ws = (char*)d_ws;
    __bf16* Wb  = (__bf16*)ws;                           // 4*TC bf16 (transposed)
    __bf16* Xb  = (__bf16*)(ws + 4 * TC * 2);            // 3*TC bf16
    float*  Kf  = (float*)(ws + 4 * TC * 2 + 3 * TC * 2);
    float*  Vf  = Kf + TC;
    float*  Rf  = Vf + TC;
    float*  Pa  = Rf + TC;
    float*  Pb  = Pa + (size_t)Cdim * NCH;
    float*  Sa  = Pb + (size_t)Cdim * NCH;
    float*  Sb  = Sa + (size_t)Cdim * NCH;
    __bf16* rwkvb = Xb;          // reuse xk slot (consumed by first GEMM)

    const int TCblocks = (int)((TC + 255) / 256);

    prep_mix<<<TCblocks, 256, 0, stream>>>(x, tmk, tmv, tmr,
                                           Xb, Xb + TC, Xb + 2 * TC);
    cvt_w_t<<<dim3(TCblocks, 4), 256, 0, stream>>>(Wk, Wv, Wr, Wo, Wb);

    dim3 ggrid(GN / BN, GM / BM);
    gemm_bf16<<<ggrid, 256, 0, stream>>>(Xb,          Wb,          Kf);
    gemm_bf16<<<ggrid, 256, 0, stream>>>(Xb + TC,     Wb + TC,     Vf);
    gemm_bf16<<<ggrid, 256, 0, stream>>>(Xb + 2 * TC, Wb + 2 * TC, Rf);

    wkv_partial<<<dim3(Cdim / 256, NCH), 256, 0, stream>>>(Kf, Vf, td, Pa, Pb);
    wkv_scan<<<Cdim / 256, 256, 0, stream>>>(Pa, Pb, td, Sa, Sb);
    wkv_out<<<dim3(Cdim / 256, NCH), 256, 0, stream>>>(Kf, Vf, Rf, td, tf,
                                                       Sa, Sb, rwkvb);

    gemm_bf16<<<ggrid, 256, 0, stream>>>(rwkvb, Wb + 3 * TC, (float*)d_out);
}
